// SelfAttention_50233937494536
// MI455X (gfx1250) — compile-verified, matrix operand order checked by reading, not run on hardware
//
#include <hip/hip_runtime.h>
#include <math.h>

// ---------------------------------------------------------------------------
// Causal self-attention forward for MI455X (gfx1250, wave32, WMMA).
//   B=4, T=2048, C=1024.  ~120 GFLOP -> compute bound -> bf16 WMMA w/ f32 acc.
// Pipeline:
//   K1: qkv = x @ W^T + b           (f32 in, bf16 out, Q pre-scaled by 1/(32+1e-5))
//   K2: S[b] = Q[b] @ K[b]^T        (bf16 out, [2048,2048] per batch)
//   K3: causal softmax rows of S    (f32 math, in-place bf16)
//   K4: O[b] = P[b] @ V[b]          (f32 out -> d_out)
// Workspace: qkv bf16 (48MB) + S bf16 (32MB) = ~84MB in d_ws.
//
// GEMM: 256 thr (8 wave32), block tile 128x128, wave tile 32x64
// (8 wmma per 32-wide K chunk from 2 A + 4 B fragments), double-buffered LDS.
// ---------------------------------------------------------------------------

typedef __bf16 bf16_t;
typedef __attribute__((ext_vector_type(16))) __bf16 v16bf;
typedef __attribute__((ext_vector_type(8)))  float  v8f;

union Frag { v16bf v; uint4 q[2]; };

// LDS strides (bf16 elements). 40 -> 80B rows, keeps 16B alignment for the
// per-lane fragment b128 reads (K-base byte offsets 0/16/32/48).
constexpr int AS_STRIDE  = 40;   // As[128][40]
constexpr int BS_STRIDE  = 40;   // Bs[128][40]   (B from [N,K] source)
constexpr int BS2_STRIDE = 136;  // Bs[32][136]   (B from [K,N] source), 272B rows

// Generic tiled GEMM:  C[M,N] = A[M,K] * B  (+bias) with B source either
// row-major [N,K] (BTRANS=true, i.e. C = A * Bsrc^T) or row-major [K,N].
// EPI: 0 = bf16 out, 1 = bf16 out + bias + qscale on cols<1024, 2 = f32 out.
template <typename AT, typename BTy, bool BTRANS, int EPI>
__global__ __launch_bounds__(256) void gemm_wmma(
    const AT* __restrict__ A, long lda, long aBatch,
    const BTy* __restrict__ Bsrc, long ldb, long bBatch,
    const float* __restrict__ bias,
    void* __restrict__ Cout, long ldc, long cBatch,
    int Kdim, float qscale)
{
    __shared__ alignas(16) bf16_t AsBuf[2][128 * AS_STRIDE];   // 2 x 10240 B
    __shared__ alignas(16) bf16_t BsBuf[2][128 * BS_STRIDE];   // 2 x 10240 B

    A    += (long)blockIdx.z * aBatch;
    Bsrc += (long)blockIdx.z * bBatch;

    const long tileM0 = (long)blockIdx.y * 128;
    const long tileN0 = (long)blockIdx.x * 128;

    const int t     = threadIdx.x;
    const int lane  = t & 31;
    const int wave  = t >> 5;
    const int mw    = wave >> 1;   // 0..3 : which 32-row strip
    const int nw    = wave & 1;    // 0..1 : which 64-col strip
    const int l15   = lane & 15;
    const int lhalf = lane >> 4;   // 0 / 1
    const int kb    = lhalf * 8;   // fragment K base per ISA 16-bit layout

    // ---- staging helpers (128x32 tiles, 16 elems per thread) ----
    auto stageA = [&](int kc, bf16_t* As) {
        const int r  = t >> 1;
        const int k0 = (t & 1) * 16;
        const AT* src = A + (tileM0 + r) * lda + kc + k0;
        bf16_t*   dst = As + r * AS_STRIDE + k0;
        if constexpr (sizeof(AT) == 4) {
#pragma unroll
            for (int i = 0; i < 16; i += 4) {
                const float4 f = *(const float4*)(src + i);
                dst[i]   = (bf16_t)f.x; dst[i+1] = (bf16_t)f.y;
                dst[i+2] = (bf16_t)f.z; dst[i+3] = (bf16_t)f.w;
            }
        } else {
            *(uint4*)dst       = *(const uint4*)src;
            *(uint4*)(dst + 8) = *(const uint4*)(src + 8);
        }
    };
    auto stageB = [&](int kc, bf16_t* Bs) {
        if constexpr (BTRANS) {           // Bsrc row-major [N,K]: Bs[n][k]
            const int n  = t >> 1;
            const int k0 = (t & 1) * 16;
            const BTy* src = Bsrc + (tileN0 + n) * ldb + kc + k0;
            bf16_t*    dst = Bs + n * BS_STRIDE + k0;
            if constexpr (sizeof(BTy) == 4) {
#pragma unroll
                for (int i = 0; i < 16; i += 4) {
                    const float4 f = *(const float4*)(src + i);
                    dst[i]   = (bf16_t)f.x; dst[i+1] = (bf16_t)f.y;
                    dst[i+2] = (bf16_t)f.z; dst[i+3] = (bf16_t)f.w;
                }
            } else {
                *(uint4*)dst       = *(const uint4*)src;
                *(uint4*)(dst + 8) = *(const uint4*)(src + 8);
            }
        } else {                           // Bsrc row-major [K,N]: Bs[k][n]
            const int k  = t >> 3;
            const int n0 = (t & 7) * 16;
            const BTy* src = Bsrc + (long)(kc + k) * ldb + tileN0 + n0;
            bf16_t*    dst = Bs + k * BS2_STRIDE + n0;
            *(uint4*)dst       = *(const uint4*)src;
            *(uint4*)(dst + 8) = *(const uint4*)(src + 8);
        }
    };

    v8f acc[2][4] = {};

    stageA(0, AsBuf[0]);
    stageB(0, BsBuf[0]);
    __syncthreads();

    int cur = 0;
    for (int kc = 0; kc < Kdim; kc += 32) {
        const int nxt = cur ^ 1;
        // Prefetch next chunk into the other buffer (overlaps with WMMAs).
        if (kc + 32 < Kdim) {
            stageA(kc + 32, AsBuf[nxt]);
            stageB(kc + 32, BsBuf[nxt]);
        }

        const bf16_t* As = AsBuf[cur];
        const bf16_t* Bs = BsBuf[cur];

        Frag af[2];
#pragma unroll
        for (int i = 0; i < 2; ++i) {
            const bf16_t* p = As + (mw * 32 + i * 16 + l15) * AS_STRIDE;
            af[i].q[0] = *(const uint4*)(p + kb);        // K = kb .. kb+7
            af[i].q[1] = *(const uint4*)(p + 16 + kb);   // K = 16+kb .. +7
        }
#pragma unroll
        for (int j = 0; j < 4; ++j) {
            Frag bfg;
            const int n = (nw * 4 + j) * 16 + l15;
            if constexpr (BTRANS) {
                const bf16_t* p = Bs + n * BS_STRIDE;
                bfg.q[0] = *(const uint4*)(p + kb);
                bfg.q[1] = *(const uint4*)(p + 16 + kb);
            } else {
#pragma unroll
                for (int e = 0; e < 16; ++e) {
                    const int kk = ((e & 8) << 1) + kb + (e & 7);
                    bfg.v[e] = Bs[kk * BS2_STRIDE + n];
                }
            }
#pragma unroll
            for (int i = 0; i < 2; ++i) {
                acc[i][j] = __builtin_amdgcn_wmma_f32_16x16x32_bf16(
                    false, af[i].v, false, bfg.v, (short)0, acc[i][j], false, false);
            }
        }
        __syncthreads();   // staging of nxt done; everyone done reading cur
        cur = nxt;
    }

    // ---- epilogue: C/D layout -> VGPR r holds M = r + 8*lhalf, N = l15 ----
    const long cBase = (long)blockIdx.z * cBatch;
#pragma unroll
    for (int i = 0; i < 2; ++i) {
#pragma unroll
        for (int j = 0; j < 4; ++j) {
            const long gcol = tileN0 + (nw * 4 + j) * 16 + l15;
#pragma unroll
            for (int r = 0; r < 8; ++r) {
                const long grow = tileM0 + mw * 32 + i * 16 + r + lhalf * 8;
                float v = acc[i][j][r];
                if constexpr (EPI == 1) {
                    v += bias[gcol];
                    if (gcol < 1024) v *= qscale;        // fold attn scale into Q
                    ((bf16_t*)Cout)[cBase + grow * ldc + gcol] = (bf16_t)v;
                } else if constexpr (EPI == 0) {
                    ((bf16_t*)Cout)[cBase + grow * ldc + gcol] = (bf16_t)v;
                } else {
                    ((float*)Cout)[cBase + grow * ldc + gcol] = v;
                }
            }
        }
    }
}

// Causal row softmax, in place on bf16 S. One block per row, 256 thr x 8 elems.
__global__ __launch_bounds__(256) void softmax_causal_kernel(bf16_t* __restrict__ S, int T)
{
    __shared__ float red[8];
    const int  trow = blockIdx.x;
    bf16_t* row = S + ((long)blockIdx.y * T + trow) * (long)T;
    const int s0 = threadIdx.x * 8;

    union { uint4 q; bf16_t h[8]; } u;
    u.q = *(const uint4*)(row + s0);

    float v[8];
#pragma unroll
    for (int i = 0; i < 8; ++i)
        v[i] = (s0 + i <= trow) ? (float)u.h[i] : -__builtin_inff();

    float m = v[0];
#pragma unroll
    for (int i = 1; i < 8; ++i) m = fmaxf(m, v[i]);
    for (int off = 16; off; off >>= 1) m = fmaxf(m, __shfl_xor(m, off, 32));
    if ((threadIdx.x & 31) == 0) red[threadIdx.x >> 5] = m;
    __syncthreads();
    m = red[0];
#pragma unroll
    for (int w = 1; w < 8; ++w) m = fmaxf(m, red[w]);
    __syncthreads();

    float p[8], sum = 0.f;
#pragma unroll
    for (int i = 0; i < 8; ++i) { p[i] = __expf(v[i] - m); sum += p[i]; }
    for (int off = 16; off; off >>= 1) sum += __shfl_xor(sum, off, 32);
    if ((threadIdx.x & 31) == 0) red[threadIdx.x >> 5] = sum;
    __syncthreads();
    sum = 0.f;
#pragma unroll
    for (int w = 0; w < 8; ++w) sum += red[w];

    const float inv = 1.0f / sum;   // diagonal element always valid -> sum >= 1
#pragma unroll
    for (int i = 0; i < 8; ++i) u.h[i] = (bf16_t)(p[i] * inv);
    *(uint4*)(row + s0) = u.q;
}

extern "C" void kernel_launch(void* const* d_in, const int* in_sizes, int n_in,
                              void* d_out, int out_size, void* d_ws, size_t ws_size,
                              hipStream_t stream)
{
    (void)in_sizes; (void)n_in; (void)out_size; (void)ws_size;
    const float* x = (const float*)d_in[0];   // [4,2048,1024]
    const float* W = (const float*)d_in[1];   // [3072,1024]
    const float* b = (const float*)d_in[2];   // [3072]
    float* out = (float*)d_out;               // [4,2048,1024] f32

    const long BT3C = (long)8192 * 3072;      // qkv elements
    bf16_t* qkv = (bf16_t*)d_ws;                               // 48 MB
    bf16_t* S   = (bf16_t*)((char*)d_ws + (size_t)BT3C * 2);   // 32 MB

    const float qscale = 1.0f / (32.0f + 1e-5f);  // 1/(sqrt(1024)+1e-5)
    dim3 blk(256);

    // K1: qkv = x @ W^T + b   (M=8192, N=3072, K=1024)
    gemm_wmma<float, float, true, 1><<<dim3(24, 64, 1), blk, 0, stream>>>(
        x, 1024, 0, W, 1024, 0, b, qkv, 3072, 0, 1024, qscale);

    // K2: S[b] = Q @ K^T      (per batch: M=N=2048, K=1024)
    gemm_wmma<bf16_t, bf16_t, true, 0><<<dim3(16, 16, 4), blk, 0, stream>>>(
        qkv, 3072, (long)2048 * 3072,
        qkv + 1024, 3072, (long)2048 * 3072, nullptr,
        S, 2048, (long)2048 * 2048, 1024, 0.f);

    // K3: causal softmax rows
    softmax_causal_kernel<<<dim3(2048, 4, 1), blk, 0, stream>>>(S, 2048);

    // K4: O[b] = P @ V        (per batch: M=2048, N=1024, K=2048)
    gemm_wmma<bf16_t, bf16_t, false, 2><<<dim3(8, 16, 4), blk, 0, stream>>>(
        S, 2048, (long)2048 * 2048,
        qkv + 2048, 3072, (long)2048 * 3072, nullptr,
        out, 1024, (long)2048 * 1024, 2048, 0.f);
}
